// VectorQuantizerEMA_52140902973505
// MI455X (gfx1250) — matrix operand order repaired and checked by reading, not compiled, not run on hardware
//
#include <hip/hip_runtime.h>
#include <hip/hip_bf16.h>
#include <math.h>

// ---------------------------------------------------------------------------
// VQ-VAE EMA vector quantizer for MI455X (gfx1250, wave32, WMMA)
// N=65536 rows, D=256, K=1024 codes.
// Distance GEMM fused with argmin using v_wmma_f32_16x16x32_bf16.
// All 16 fragments for a tile are loaded into distinct registers before the
// WMMA chain so the scheduler emits partial s_wait_dscnt (overlap), and the
// next codebook tile is prefetched global->reg during compute.
// Fully deterministic: no floating-point atomics; fixed-shape reductions.
// ---------------------------------------------------------------------------

#define NROWS   65536
#define DDIM    256
#define KCODES  1024

typedef __attribute__((ext_vector_type(16))) __bf16 bf16x16;
typedef __attribute__((ext_vector_type(8)))  float  f32x8;

union FragU { bf16x16 v; uint4 u[2]; };

// Output layout (floats, concatenated in reference return order)
static constexpr size_t Q_OFF    = 1;                               // quantized_st
static constexpr size_t PERP_OFF = Q_OFF + (size_t)NROWS * DDIM;    // perplexity
static constexpr size_t EMBW_OFF = PERP_OFF + 1;                    // new_embedding_w
static constexpr size_t CS_OFF   = EMBW_OFF + (size_t)KCODES * DDIM;// new_cluster_size
static constexpr size_t EMAW_OFF = CS_OFF + KCODES;                 // new_ema_w

__device__ __forceinline__ unsigned short f2bf(float f) {
    unsigned u = __float_as_uint(f);
    unsigned r = u + 0x7FFFu + ((u >> 16) & 1u);   // round-to-nearest-even
    return (unsigned short)(r >> 16);
}

// ---------------------------------------------------------------------------
// K1: codebook -> bf16 copy + exact f32 squared norms
// grid(KCODES), block(256)
// ---------------------------------------------------------------------------
__global__ __launch_bounds__(256) void k_enorm_bf16(
    const float* __restrict__ E, unsigned short* __restrict__ Ebf,
    float* __restrict__ enorm) {
    const int k = blockIdx.x, t = threadIdx.x;
    const float f = E[(size_t)k * DDIM + t];
    Ebf[(size_t)k * DDIM + t] = f2bf(f);
    __shared__ float red[256];
    red[t] = f * f;
    __syncthreads();
    for (int s = 128; s > 0; s >>= 1) { if (t < s) red[t] += red[t + s]; __syncthreads(); }
    if (t == 0) enorm[k] = red[0];
}

// ---------------------------------------------------------------------------
// K2: fused GEMM(argmin).  block = 256 thr = 8 waves (2 M-waves x 4 N-waves)
// block tile: 32 rows x 64 codes/iter, 16 iters over K=1024.
// grid(NROWS/32)
// ---------------------------------------------------------------------------
__device__ __forceinline__ void load_tile_regs(
    uint4* stage, const unsigned short* __restrict__ Ebf, int kb, int tid) {
    #pragma unroll
    for (int i = 0; i < 8; ++i) {
        const int elem = (tid + 256 * i) * 8;       // 0..16383
        const int r = elem >> 8, c = elem & 255;
        stage[i] = *(const uint4*)(Ebf + (size_t)(kb + r) * DDIM + c);
    }
}

__global__ __launch_bounds__(256) void k_argmin(
    const float* __restrict__ x, const unsigned short* __restrict__ Ebf,
    const float* __restrict__ enorm, int* __restrict__ idxOut) {
    __shared__ __attribute__((aligned(16))) unsigned short As[32][264]; // +8 pad: 528B row stride
    __shared__ __attribute__((aligned(16))) unsigned short Bs[64][264];
    __shared__ float redD[32][4];
    __shared__ int   redI[32][4];

    const int tid  = threadIdx.x;
    const int lane = tid & 31, wave = tid >> 5;
    const int wr = wave >> 2, wc = wave & 3;      // wave row (M), wave col (N)
    const int half = lane >> 4, l15 = lane & 15;
    const int m0 = blockIdx.x * 32;

    // stage x tile as bf16 into LDS (coalesced float4 reads)
    #pragma unroll
    for (int i = 0; i < 8; ++i) {
        const int elem = (tid + 256 * i) * 4;       // 0..8191
        const int r = elem >> 8, c = elem & 255;
        const float4 v = *(const float4*)(x + (size_t)(m0 + r) * DDIM + c);
        uint2 p;
        p.x = (unsigned)f2bf(v.x) | ((unsigned)f2bf(v.y) << 16);
        p.y = (unsigned)f2bf(v.z) | ((unsigned)f2bf(v.w) << 16);
        *(uint2*)&As[r][c] = p;
    }

    float bd[8];
    int   bi[8];
    #pragma unroll
    for (int r = 0; r < 8; ++r) { bd[r] = 3.4e38f; bi[r] = 0; }

    const int arow = 16 * wr + l15;
    const int brow = wc * 16 + l15;
    const int aoff = 8 * half;     // A elems 0-7 -> K d0+8h..  , 8-15 -> d0+16+8h
    const int boff = 8 * half;     // B elems 0-7 -> K d0+8h..  , 8-15 -> d0+16+8h

    // prefetch codebook tile 0 into registers (global latency overlaps As staging)
    uint4 stage[8];
    load_tile_regs(stage, Ebf, 0, tid);

    for (int it = 0; it < 16; ++it) {
        const int kb = it * 64;
        __syncthreads();                    // Bs free (and As visible on it==0)
        // commit staged tile to LDS (s_wait_loadcnt folds in here)
        #pragma unroll
        for (int i = 0; i < 8; ++i) {
            const int elem = (tid + 256 * i) * 8;
            const int r = elem >> 8, c = elem & 255;
            *(uint4*)&Bs[r][c] = stage[i];
        }
        __syncthreads();

        // issue the ||e||^2 load FIRST, then the next-tile prefetch: in-order
        // LOADcnt lets the epilogue wait on enorm with loadcnt<=8 while the
        // 32KB prefetch stays in flight until next iteration's ds_store.
        const int code = kb + wc * 16 + l15;
        const float en = enorm[code];
        if (it < 15) load_tile_regs(stage, Ebf, kb + 64, tid);

        // load ALL 16 fragments into distinct registers -> scheduler must use
        // partial s_wait_dscnt, overlapping ds_load latency with the WMMA chain
        FragU af[8], bf_[8];
        #pragma unroll
        for (int s = 0; s < 8; ++s) {
            const int d0 = s * 32;
            af[s].u[0]  = *(const uint4*)&As[arow][d0 + aoff];
            af[s].u[1]  = *(const uint4*)&As[arow][d0 + aoff + 16];
            bf_[s].u[0] = *(const uint4*)&Bs[brow][d0 + boff];
            bf_[s].u[1] = *(const uint4*)&Bs[brow][d0 + boff + 16];
        }

        f32x8 acc;
        #pragma unroll
        for (int r = 0; r < 8; ++r) acc[r] = 0.0f;
        #pragma unroll
        for (int s = 0; s < 8; ++s) {          // D loop: 8 x K=32
            acc = __builtin_amdgcn_wmma_f32_16x16x32_bf16(
                false, af[s].v, false, bf_[s].v, (short)0, acc, false, false);
        }

        // distance = ||e||^2 - 2*dot  (||x||^2 is row-constant: irrelevant to argmin)
        #pragma unroll
        for (int r = 0; r < 8; ++r) {
            const float dist = fmaf(-2.0f, acc[r], en);
            if (dist < bd[r]) { bd[r] = dist; bi[r] = code; }   // strict < keeps lowest idx
        }
    }

    // reduce across the 16 lanes sharing the same rows (ties -> lower index)
    #pragma unroll
    for (int m = 1; m < 16; m <<= 1) {
        #pragma unroll
        for (int r = 0; r < 8; ++r) {
            const float pd = __shfl_xor(bd[r], m, 32);
            const int   pi = __shfl_xor(bi[r], m, 32);
            if (pd < bd[r] || (pd == bd[r] && pi < bi[r])) { bd[r] = pd; bi[r] = pi; }
        }
    }
    if (l15 == 0) {
        #pragma unroll
        for (int r = 0; r < 8; ++r) {
            const int ml = 16 * wr + 8 * half + r;   // C layout: M = r + 8*half
            redD[ml][wc] = bd[r];
            redI[ml][wc] = bi[r];
        }
    }
    __syncthreads();
    if (tid < 32) {                                   // reduce over 4 N-waves
        float best = redD[tid][0]; int bidx = redI[tid][0];
        for (int w = 1; w < 4; ++w) {
            const float d2 = redD[tid][w]; const int i2 = redI[tid][w];
            if (d2 < best || (d2 == best && i2 < bidx)) { best = d2; bidx = i2; }
        }
        idxOut[m0 + tid] = bidx;
    }
}

// ---------------------------------------------------------------------------
// K3a: per-chunk histogram (256 rows/chunk). grid(256), block(256).
// LDS integer atomics only -> order-independent.
// ---------------------------------------------------------------------------
__global__ __launch_bounds__(256) void k_hist(
    const int* __restrict__ idxArr, int* __restrict__ hist) {
    __shared__ int lh[KCODES];
    const int t = threadIdx.x, b = blockIdx.x;
    for (int j = t; j < KCODES; j += 256) lh[j] = 0;
    __syncthreads();
    atomicAdd(&lh[idxArr[b * 256 + t]], 1);
    __syncthreads();
    for (int j = t; j < KCODES; j += 256) hist[(size_t)b * KCODES + j] = lh[j];
}

// ---------------------------------------------------------------------------
// K3b: single block, 1024 threads (thread = code k).
// counts, exclusive code bases, per-chunk stable offsets, new_cluster_size,
// Laplace cs, perplexity. Fixed-shape trees -> deterministic.
// ---------------------------------------------------------------------------
__global__ __launch_bounds__(1024) void k_scan(
    int* __restrict__ hist, const float* __restrict__ emaCS,
    int* __restrict__ codeBase, int* __restrict__ countsI,
    float* __restrict__ csArr, float* __restrict__ out) {
    __shared__ float fr[KCODES];
    __shared__ int   sc[KCODES];
    const int k = threadIdx.x;

    int colsum = 0;
    for (int b = 0; b < 256; ++b) colsum += hist[(size_t)b * KCODES + k];
    countsI[k] = colsum;

    // inclusive Hillis-Steele scan over k
    sc[k] = colsum;
    __syncthreads();
    for (int off = 1; off < KCODES; off <<= 1) {
        int v = (k >= off) ? sc[k - off] : 0;
        __syncthreads();
        sc[k] += v;
        __syncthreads();
    }
    const int base = sc[k] - colsum;
    codeBase[k] = base;

    // convert per-chunk histogram to per-chunk stable base offsets
    int run = base;
    for (int b = 0; b < 256; ++b) {
        const int tmp = hist[(size_t)b * KCODES + k];
        hist[(size_t)b * KCODES + k] = run;
        run += tmp;
    }

    // EMA cluster size + Laplace smoothing
    const float ncs = 0.99f * emaCS[k] + 0.01f * (float)colsum;
    out[CS_OFF + k] = ncs;
    fr[k] = ncs;
    __syncthreads();
    for (int s = 512; s > 0; s >>= 1) { if (k < s) fr[k] += fr[k + s]; __syncthreads(); }
    const float n = fr[0];
    __syncthreads();
    csArr[k] = (ncs + 1e-5f) / (n + (float)KCODES * 1e-5f) * n;

    // perplexity from counts
    const float p = (float)colsum * (1.0f / (float)NROWS);
    fr[k] = p * logf(p + 1e-10f);
    __syncthreads();
    for (int s = 512; s > 0; s >>= 1) { if (k < s) fr[k] += fr[k + s]; __syncthreads(); }
    if (k == 0) out[PERP_OFF] = expf(-fr[0]);
}

// ---------------------------------------------------------------------------
// K3c: stable scatter of row ids into code-sorted permutation. grid(256).
// Sequential within chunk (thread 0) -> stable; chunks independent.
// ---------------------------------------------------------------------------
__global__ __launch_bounds__(256) void k_scatter(
    const int* __restrict__ idxArr, const int* __restrict__ hist,
    int* __restrict__ perm) {
    __shared__ int cur[KCODES];
    const int t = threadIdx.x, b = blockIdx.x;
    for (int j = t; j < KCODES; j += 256) cur[j] = hist[(size_t)b * KCODES + j];
    __syncthreads();
    if (t == 0) {
        for (int j = 0; j < 256; ++j) {
            const int m = b * 256 + j;
            const int k = idxArr[m];
            perm[cur[k]++] = m;
        }
    }
}

// ---------------------------------------------------------------------------
// K4: per-code dw sum (fixed ascending order), EMA weight + normalized
// codebook update. grid(KCODES), block(256) (thread = dim d).
// ---------------------------------------------------------------------------
__global__ __launch_bounds__(256) void k_codebook(
    const float* __restrict__ x, const float* __restrict__ emaW,
    const int* __restrict__ codeBase, const int* __restrict__ countsI,
    const float* __restrict__ csArr, const int* __restrict__ perm,
    float* __restrict__ out) {
    const int k = blockIdx.x, d = threadIdx.x;
    const int base = codeBase[k], cnt = countsI[k];
    float acc = 0.0f;
    for (int j = 0; j < cnt; ++j) {
        const int m = perm[base + j];
        acc += x[(size_t)m * DDIM + d];
    }
    const float nw = 0.99f * emaW[(size_t)k * DDIM + d] + 0.01f * acc;
    out[EMAW_OFF + (size_t)k * DDIM + d] = nw;
    out[EMBW_OFF + (size_t)k * DDIM + d] = nw / csArr[k];
}

// ---------------------------------------------------------------------------
// K5: gather quantized (== quantized_st numerically) + per-block loss partial.
// grid(NROWS/32), block(256) (thread = dim d).
// ---------------------------------------------------------------------------
__global__ __launch_bounds__(256) void k_quant(
    const float* __restrict__ x, const float* __restrict__ E,
    const int* __restrict__ idxArr, float* __restrict__ out,
    float* __restrict__ partials) {
    const int t = threadIdx.x;
    const int m0 = blockIdx.x * 32;
    float local = 0.0f;
    for (int r = 0; r < 32; ++r) {
        const int m = m0 + r;
        const int k = idxArr[m];
        const float q  = E[(size_t)k * DDIM + t];
        const float xv = x[(size_t)m * DDIM + t];
        out[Q_OFF + (size_t)m * DDIM + t] = q;   // x + sg(q - x) == q
        const float df = q - xv;
        local += df * df;
    }
    __shared__ float red[256];
    red[t] = local;
    __syncthreads();
    for (int s = 128; s > 0; s >>= 1) { if (t < s) red[t] += red[t + s]; __syncthreads(); }
    if (t == 0) partials[blockIdx.x] = red[0];
}

// K6: final deterministic loss reduction over 2048 partials. grid(1).
__global__ __launch_bounds__(256) void k_loss(
    const float* __restrict__ partials, float* __restrict__ out) {
    const int t = threadIdx.x;
    float s = 0.0f;
    for (int i = 0; i < 8; ++i) s += partials[t + 256 * i];
    __shared__ float red[256];
    red[t] = s;
    __syncthreads();
    for (int st = 128; st > 0; st >>= 1) { if (t < st) red[t] += red[t + st]; __syncthreads(); }
    if (t == 0) out[0] = 0.25f * red[0] / ((float)NROWS * (float)DDIM);
}

// ---------------------------------------------------------------------------
extern "C" void kernel_launch(void* const* d_in, const int* in_sizes, int n_in,
                              void* d_out, int out_size, void* d_ws, size_t ws_size,
                              hipStream_t stream) {
    (void)in_sizes; (void)n_in; (void)out_size; (void)ws_size;
    const float* x     = (const float*)d_in[0];   // [32,2048,256]
    const float* E     = (const float*)d_in[1];   // [1024,256]
    const float* emaCS = (const float*)d_in[2];   // [1024]
    const float* emaW  = (const float*)d_in[3];   // [1024,256]
    float* out = (float*)d_out;

    // ws layout (all 16B aligned)
    char* w = (char*)d_ws;
    unsigned short* Ebf   = (unsigned short*)(w);            // 524288 B
    float* enorm          = (float*)(w + 524288);            //   4096 B
    int*   idxArr         = (int*)  (w + 528384);            // 262144 B
    int*   hist           = (int*)  (w + 790528);            // 1048576 B
    int*   codeBase       = (int*)  (w + 1839104);           //   4096 B
    int*   countsI        = (int*)  (w + 1843200);           //   4096 B
    float* csArr          = (float*)(w + 1847296);           //   4096 B
    int*   perm           = (int*)  (w + 1851392);           // 262144 B
    float* partials       = (float*)(w + 2113536);           //   8192 B

    k_enorm_bf16<<<dim3(KCODES),      dim3(256),  0, stream>>>(E, Ebf, enorm);
    k_argmin    <<<dim3(NROWS / 32),  dim3(256),  0, stream>>>(x, Ebf, enorm, idxArr);
    k_hist      <<<dim3(256),         dim3(256),  0, stream>>>(idxArr, hist);
    k_scan      <<<dim3(1),           dim3(1024), 0, stream>>>(hist, emaCS, codeBase,
                                                               countsI, csArr, out);
    k_scatter   <<<dim3(256),         dim3(256),  0, stream>>>(idxArr, hist, perm);
    k_codebook  <<<dim3(KCODES),      dim3(256),  0, stream>>>(x, emaW, codeBase, countsI,
                                                               csArr, perm, out);
    k_quant     <<<dim3(NROWS / 32),  dim3(256),  0, stream>>>(x, E, idxArr, out, partials);
    k_loss      <<<dim3(1),           dim3(256),  0, stream>>>(partials, out);
}
